// RITSI_25958782337188
// MI455X (gfx1250) — matrix operand order repaired and checked by reading, not compiled, not run on hardware
//
#include <hip/hip_runtime.h>

#define BATCH 256
#define TLEN  2048
#define ISZ   64
#define HSZ   32
#define NCLS  2

typedef __attribute__((ext_vector_type(16))) _Float16 v16h;
typedef __attribute__((ext_vector_type(8)))  float    v8f;

// D = A(16x32 f16) * B(32x16 f16) + C(16x16 f32)
__device__ __forceinline__ v8f wmma16(v16h a, v16h b, v8f c) {
  return __builtin_amdgcn_wmma_f32_16x16x32_f16(false, a, false, b, (short)0, c, false, false);
}

__device__ __forceinline__ v8f splat8(float x) {
  v8f v = {x, x, x, x, x, x, x, x};
  return v;
}

__device__ __forceinline__ float sigm(float x) { return 1.0f / (1.0f + __expf(-x)); }
__device__ __forceinline__ float tanh_fast(float x) {
  float e = __expf(2.0f * x);
  return (e - 1.0f) / (e + 1.0f);
}

// ---- CDNA5 async global->LDS copy (ASYNCcnt-tracked), ISA 15.18.3 op 98 ----
// vdst = per-lane LDS byte address, vaddr = per-lane 64-bit global address.
__device__ __forceinline__ void async_ld_b128(void* lds_dst, const void* gsrc) {
  unsigned lds = (unsigned)(size_t)lds_dst;  // low 32 bits of flat LDS addr = LDS offset
  asm volatile("global_load_async_to_lds_b128 %0, %1, off"
               :: "v"(lds), "v"(gsrc) : "memory");
}
__device__ __forceinline__ void wait_async0() {
  asm volatile("s_wait_asynccnt 0x0" ::: "memory");
}
__device__ __forceinline__ void wait_ds0() {
  asm volatile("s_wait_dscnt 0x0" ::: "memory");
}

// ---- Fragment layout helpers (wave32, 16x16x32 f16 WMMA) ----
// B fragment (32x16, K x N): lane L holds column n = L%16; halves h=0..7 hold k = 32*ks + 8*(L/16) + h,
// h=8..15 -> k = 32*ks + 16 + 8*(L/16) + (h-8). One fragment = 512 halves (1KB), lane stride 32B.
// A fragment (16x32, M x K): lane L holds row m = L%16; halves: h=0..7 ->
// k = 8*(L/16)+h, h=8..15 -> k = 16 + 8*(L/16) + (h-8).

// Pack W^T (stored as src[n*K + k], i.e. src is [N,K] row-major) into B-fragment order.
__device__ void pack_B(_Float16* dst, const float* src, int K, int nN, int lane) {
  const int nK = K >> 5;
  for (int ks = 0; ks < nK; ++ks)
    for (int nt = 0; nt < nN; ++nt) {
      _Float16* f = dst + (ks * nN + nt) * 512;
      const int n  = nt * 16 + (lane & 15);
      const int kb = ks * 32 + ((lane >> 4) << 3);
      #pragma unroll
      for (int h = 0; h < 16; ++h)
        f[lane * 16 + h] = (_Float16)src[n * K + kb + h + ((h >> 3) << 3)];
    }
}

__device__ __forceinline__ v16h ldb(const _Float16* base, int frag, int lane) {
  return *(const v16h*)(base + frag * 512 + lane * 16);
}

// Gather an A fragment (K-slice ks) from a row-major f32 LDS tile.
__device__ __forceinline__ v16h a_from_tile(const float* tile, int ld, int ks, int lane) {
  const int m  = lane & 15;
  const int k0 = ks * 32 + ((lane >> 4) << 3);
  const float* p = tile + m * ld + k0;
  v16h a;
  #pragma unroll
  for (int j = 0; j < 8; ++j) {
    a[j]     = (_Float16)p[j];
    a[j + 8] = (_Float16)p[j + 16];
  }
  return a;
}

// Scatter-store one logical A element (m, k) into an A-fragment region in LDS.
__device__ __forceinline__ void a_store(_Float16* region, int m, int k, float v) {
  const int ks = k >> 5, kk = k & 31;
  const int lp = ((kk >> 3) & 1) << 4;
  const int hp = ((kk >> 4) << 3) | (kk & 7);
  region[ks * 512 + (m + lp) * 16 + hp] = (_Float16)v;
}

// LDS byte offsets
#define OFF_WD    0        // 4 frags  (K=64,N=32)   4KB
#define OFF_WR    4096     // 4 frags  (K=32,N=64)   4KB
#define OFF_WIH   8192     // 32 frags (K=128,N=128) 32KB
#define OFF_WHH   40960    // 8 frags  (K=32,N=128)  8KB
#define OFF_STAGE 49152    // 2 x (X 4KB | M 4KB | D 4KB) = 24KB
#define STAGE_SZ  12288
#define OFF_HA    73728    // 16x32 f16 A-frag       1KB
#define OFF_XCA   74752    // 16x64 f16 A-frags      2KB
#define SMEM_SZ   76800

__global__ void __launch_bounds__(32)
rits_lstm_kernel(const float* __restrict__ values, const float* __restrict__ masks,
                 const float* __restrict__ deltas,
                 const float* __restrict__ W_decay, const float* __restrict__ b_decay,
                 const float* __restrict__ W_reg,   const float* __restrict__ b_reg,
                 const float* __restrict__ W_ih,    const float* __restrict__ b_ih,
                 const float* __restrict__ W_hh,    const float* __restrict__ b_hh,
                 const float* __restrict__ W_out,   const float* __restrict__ b_out,
                 float* __restrict__ y_out, float* __restrict__ imp_out)
{
  __shared__ __align__(64) unsigned char smem[SMEM_SZ];
  _Float16* wdB  = (_Float16*)(smem + OFF_WD);
  _Float16* wrB  = (_Float16*)(smem + OFF_WR);
  _Float16* wihB = (_Float16*)(smem + OFF_WIH);
  _Float16* whhB = (_Float16*)(smem + OFF_WHH);
  _Float16* hAr  = (_Float16*)(smem + OFF_HA);
  _Float16* xcAr = (_Float16*)(smem + OFF_XCA);

  const int lane = threadIdx.x & 31;
  const int b0   = blockIdx.x * 16;
  const int cn   = lane & 15;          // C-layout column
  const int mrow = (lane >> 4) * 8;    // C-layout row base

  // Per-lane staging geometry: lane handles 8 float4 chunks (f4 = q*32+lane).
  // Issue async global->LDS copies of x/m/d tiles for step t into buffer buf.
  auto stage_async = [&](int t, int buf) {
    float* bx = (float*)(smem + OFF_STAGE + buf * STAGE_SZ);
    float* bm = bx + 1024;
    float* bd = bm + 1024;
    #pragma unroll
    for (int q = 0; q < 8; ++q) {
      const int f4 = q * 32 + lane;
      const int m  = f4 >> 4;
      const int o  = (f4 & 15) << 2;
      const size_t g = ((size_t)(b0 + m) * TLEN + t) * ISZ + o;
      async_ld_b128(bx + m * 64 + o, values + g);
      async_ld_b128(bm + m * 64 + o, masks  + g);
      async_ld_b128(bd + m * 64 + o, deltas + g);
      if ((f4 & 15) == 0 && t + 1 < TLEN) {  // prefetch one more step ahead
        __builtin_prefetch(values + g + ISZ, 0, 3);
        __builtin_prefetch(masks  + g + ISZ, 0, 3);
        __builtin_prefetch(deltas + g + ISZ, 0, 3);
      }
    }
  };

  // One-time: weights -> f16 B-fragment order in LDS (each lane fills its own slots).
  pack_B(wdB,  W_decay, 64, 2, lane);
  pack_B(wrB,  W_reg,   32, 4, lane);
  pack_B(wihB, W_ih,   128, 8, lane);
  pack_B(whhB, W_hh,    32, 8, lane);

  // Per-N-tile biases (C layout: value depends only on column).
  float dbias[2], rbias[4], gbias[8];
  #pragma unroll
  for (int j = 0; j < 2; ++j) dbias[j] = b_decay[j * 16 + cn];
  #pragma unroll
  for (int j = 0; j < 4; ++j) rbias[j] = b_reg[j * 16 + cn];
  #pragma unroll
  for (int j = 0; j < 8; ++j) gbias[j] = b_ih[j * 16 + cn] + b_hh[j * 16 + cn];

  float cst[16], hst[16];  // c, h state in C layout: [jtile*8 + r]
  #pragma unroll
  for (int i = 0; i < 16; ++i) { cst[i] = 0.0f; hst[i] = 0.0f; }

  stage_async(0, 0);  // prologue fill

  #pragma unroll 1
  for (int t = 0; t < TLEN; ++t) {
    const int bufc = t & 1;
    float* ldsX = (float*)(smem + OFF_STAGE + bufc * STAGE_SZ);
    float* ldsM = ldsX + 1024;
    float* ldsD = ldsM + 1024;

    wait_async0();            // buffer[t&1] fully written by TDM/async path
    wait_ds0();               // close WAR window on buffer[(t+1)&1]
    if (t + 1 < TLEN) stage_async(t + 1, bufc ^ 1);

    // ---- gamma = exp(-relu(d @ Wd^T + b)); h *= gamma ----
    v16h dA0 = a_from_tile(ldsD, 64, 0, lane);
    v16h dA1 = a_from_tile(ldsD, 64, 1, lane);
    #pragma unroll
    for (int j = 0; j < 2; ++j) {
      v8f c = splat8(dbias[j]);
      c = wmma16(dA0, ldb(wdB, 0 * 2 + j, lane), c);
      c = wmma16(dA1, ldb(wdB, 1 * 2 + j, lane), c);
      #pragma unroll
      for (int r = 0; r < 8; ++r)
        hst[j * 8 + r] *= __expf(-fmaxf(c[r], 0.0f));
    }

    // ---- transpose decayed h (C layout) -> A fragment via LDS ----
    #pragma unroll
    for (int j = 0; j < 2; ++j)
      #pragma unroll
      for (int r = 0; r < 8; ++r)
        a_store(hAr, mrow + r, j * 16 + cn, hst[j * 8 + r]);
    v16h hAf = *(const v16h*)(hAr + lane * 16);

    // ---- x_h = h @ Wreg^T + b_reg  (16x64, 4 N tiles) ----
    float xh[32];
    #pragma unroll
    for (int nt = 0; nt < 4; ++nt) {
      v8f c = splat8(rbias[nt]);
      c = wmma16(hAf, ldb(wrB, nt, lane), c);
      #pragma unroll
      for (int r = 0; r < 8; ++r) xh[nt * 8 + r] = c[r];
    }

    // ---- x_c = m*x + (1-m)*x_h : write imputations + A fragments ----
    #pragma unroll
    for (int nt = 0; nt < 4; ++nt) {
      const int i = nt * 16 + cn;
      #pragma unroll
      for (int r = 0; r < 8; ++r) {
        const int m = mrow + r;
        const float xv = ldsX[m * 64 + i];
        const float mv = ldsM[m * 64 + i];
        const float xc = mv * xv + (1.0f - mv) * xh[nt * 8 + r];
        imp_out[((size_t)(b0 + m) * TLEN + t) * ISZ + i] = xc;
        a_store(xcAr, m, i, xc);
      }
    }
    v16h xcA0 = *(const v16h*)(xcAr + lane * 16);
    v16h xcA1 = *(const v16h*)(xcAr + 512 + lane * 16);
    v16h mA0  = a_from_tile(ldsM, 64, 0, lane);
    v16h mA1  = a_from_tile(ldsM, 64, 1, lane);

    // ---- gates = [x_c, m] @ Wih^T + h @ Whh^T + (b_ih + b_hh) ----
    float G[64];
    #pragma unroll
    for (int nt = 0; nt < 8; ++nt) {
      v8f c = splat8(gbias[nt]);
      c = wmma16(xcA0, ldb(wihB, 0 * 8 + nt, lane), c);
      c = wmma16(xcA1, ldb(wihB, 1 * 8 + nt, lane), c);
      c = wmma16(mA0,  ldb(wihB, 2 * 8 + nt, lane), c);
      c = wmma16(mA1,  ldb(wihB, 3 * 8 + nt, lane), c);
      c = wmma16(hAf,  ldb(whhB, nt, lane), c);
      #pragma unroll
      for (int r = 0; r < 8; ++r) G[nt * 8 + r] = c[r];
    }

    // ---- LSTM cell update (gate tiles: 0,1=i  2,3=f  4,5=g  6,7=o) ----
    #pragma unroll
    for (int j = 0; j < 2; ++j)
      #pragma unroll
      for (int r = 0; r < 8; ++r) {
        const float ig = sigm(G[(0 + j) * 8 + r]);
        const float fg = sigm(G[(2 + j) * 8 + r]);
        const float gg = tanh_fast(G[(4 + j) * 8 + r]);
        const float og = sigm(G[(6 + j) * 8 + r]);
        const float cnew = fg * cst[j * 8 + r] + ig * gg;
        cst[j * 8 + r] = cnew;
        hst[j * 8 + r] = og * tanh_fast(cnew);
      }
  }

  // ---- y = h @ Wout^T + b_out (tiny: do via LDS row-major h) ----
  float* ldsH = (float*)(smem + OFF_STAGE);  // reuse staging
  #pragma unroll
  for (int j = 0; j < 2; ++j)
    #pragma unroll
    for (int r = 0; r < 8; ++r)
      ldsH[(mrow + r) * 32 + j * 16 + cn] = hst[j * 8 + r];
  if (lane < 16) {
    const int m = lane;
    float a0 = b_out[0], a1 = b_out[1];
    #pragma unroll
    for (int k = 0; k < 32; ++k) {
      const float hv = ldsH[m * 32 + k];
      a0 += hv * W_out[k];
      a1 += hv * W_out[32 + k];
    }
    y_out[(b0 + m) * 2 + 0] = a0;
    y_out[(b0 + m) * 2 + 1] = a1;
  }
}

extern "C" void kernel_launch(void* const* d_in, const int* in_sizes, int n_in,
                              void* d_out, int out_size, void* d_ws, size_t ws_size,
                              hipStream_t stream) {
  const float* values  = (const float*)d_in[0];
  const float* masks   = (const float*)d_in[1];
  const float* deltas  = (const float*)d_in[2];
  const float* W_decay = (const float*)d_in[3];
  const float* b_decay = (const float*)d_in[4];
  const float* W_reg   = (const float*)d_in[5];
  const float* b_reg   = (const float*)d_in[6];
  const float* W_ih    = (const float*)d_in[7];
  const float* b_ih    = (const float*)d_in[8];
  const float* W_hh    = (const float*)d_in[9];
  const float* b_hh    = (const float*)d_in[10];
  const float* W_out   = (const float*)d_in[11];
  const float* b_out   = (const float*)d_in[12];

  float* y_out   = (float*)d_out;                 // [B, C] first
  float* imp_out = (float*)d_out + BATCH * NCLS;  // then [B, T, I]

  hipLaunchKernelGGL(rits_lstm_kernel, dim3(BATCH / 16), dim3(32), 0, stream,
                     values, masks, deltas, W_decay, b_decay, W_reg, b_reg,
                     W_ih, b_ih, W_hh, b_hh, W_out, b_out, y_out, imp_out);
}
